// GATModelWithAttention_45011257262711
// MI455X (gfx1250) — matrix operand order repaired and checked by reading, not compiled, not run on hardware
//
#include <hip/hip_runtime.h>
#include <hip/hip_bf16.h>

// ---------------------------------------------------------------------------
// GATModelWithAttention — MI455X (gfx1250) optimized forward.
//
// Key algebraic fact: the cross-attention has query-len == key-len == 1, so
// softmax over the singleton key axis is exactly 1.0 and ctx == V.  Q (and
// therefore nf, the pooling, and all three GATv2 layers) cannot influence the
// logits.  Live computation:
//   V   = biobert_cls @ ca_wv + ca_bv            [256,256]  (K=768)
//   P   = V @ ca_wo + ca_bo                      [256,256]  (K=256)
//   ctx = LN(P; ca_ln_g, ca_ln_b)
//   T   = relu(ctx @ ffn_w1 + ffn_b1)            [256,256]
//   F   = T @ ffn_w2 + ffn_b2                    [256,256]
//   out = LN(ctx + F; fn_g, fn_b)
//   logits = out @ fc_w + fc_b                   [256,5]
// ~0.2 GFLOP, a few MB of traffic: launch-latency bound on MI455X.
// GEMMs use v_wmma_f32_16x16x32_f16 (wave32, one 16x16 tile per wave).
// ---------------------------------------------------------------------------

typedef __attribute__((ext_vector_type(16))) _Float16 v16h;
typedef __attribute__((ext_vector_type(8)))  float    v8f;

// C = act(A[M,K] @ B[K,N] + bias[N]).  One wave (32 threads) per 16x16 tile.
// Fragment layouts used here (contiguous k order, same for A and B):
//   A 16x32 f16 : lane = h*16 + m (h=lane/16, m=lane%16); element e in [0,16):
//                 k = 16*h + e
//   B 32x16 f16 : n = lane%16; element e: k = 16*h + e
//   C/D 16x16 f32: n = lane%16; acc[i] = C[8*h + i][n]
__global__ void wmma_gemm_f16(const float* __restrict__ A,
                              const float* __restrict__ B,
                              const float* __restrict__ bias,
                              float* __restrict__ C,
                              int M, int N, int K, int relu)
{
    const int tilesN = N >> 4;
    const int mt     = blockIdx.x / tilesN;
    const int nt     = blockIdx.x % tilesN;
    const int lane   = threadIdx.x & 31;
    const int half   = lane >> 4;   // 0..1
    const int l16    = lane & 15;   // m (for A) / n (for B,C)

    const int mBase = mt << 4;
    const int col   = (nt << 4) + l16;

    v8f acc = {};
    for (int k0 = 0; k0 < K; k0 += 32) {
        v16h a, b;
        const float* arow = A + (size_t)(mBase + l16) * K + k0 + half * 16;
        #pragma unroll
        for (int g = 0; g < 2; ++g)
            #pragma unroll
            for (int w = 0; w < 8; ++w)
                a[8 * g + w] = (_Float16)arow[8 * g + w];

        const float* bcol = B + (size_t)(k0 + half * 16) * N + col;
        #pragma unroll
        for (int e = 0; e < 16; ++e)
            b[e] = (_Float16)bcol[(size_t)e * N];

        // 8 args: (neg_a, A, neg_b, B, c_mod, C, reuse_a, reuse_b)
        acc = __builtin_amdgcn_wmma_f32_16x16x32_f16(
            false, a, false, b, (short)0, acc, false, false);
    }

    const float bs = bias ? bias[col] : 0.0f;
    #pragma unroll
    for (int i = 0; i < 8; ++i) {
        float v = acc[i] + bs;
        if (relu) v = v > 0.0f ? v : 0.0f;
        C[(size_t)(mBase + half * 8 + i) * N + col] = v;
    }
}

// Row LayerNorm over D=256 (optionally with residual R added first).
// One wave32 per row, 8 contiguous floats per lane, shfl_xor butterfly reduce.
__global__ void layernorm256(const float* __restrict__ X,
                             const float* __restrict__ R,
                             const float* __restrict__ g,
                             const float* __restrict__ b,
                             float* __restrict__ Y)
{
    const int row  = blockIdx.x;
    const int lane = threadIdx.x & 31;
    const int cb   = lane * 8;

    float v[8];
    float s = 0.0f;
    const float* xp = X + (size_t)row * 256 + cb;
    const float* rp = R ? R + (size_t)row * 256 + cb : nullptr;
    #pragma unroll
    for (int i = 0; i < 8; ++i) {
        float t = xp[i];
        if (rp) t += rp[i];
        v[i] = t;
        s += t;
    }
    #pragma unroll
    for (int off = 16; off > 0; off >>= 1) s += __shfl_xor(s, off, 32);
    const float mean = s * (1.0f / 256.0f);

    float q = 0.0f;
    #pragma unroll
    for (int i = 0; i < 8; ++i) { float d = v[i] - mean; q += d * d; }
    #pragma unroll
    for (int off = 16; off > 0; off >>= 1) q += __shfl_xor(q, off, 32);
    const float inv = rsqrtf(q * (1.0f / 256.0f) + 1e-5f);

    float* yp = Y + (size_t)row * 256 + cb;
    #pragma unroll
    for (int i = 0; i < 8; ++i)
        yp[i] = (v[i] - mean) * inv * g[cb + i] + b[cb + i];
}

// logits[256,5] = X[256,256] @ W[256,5] + b[5] — too thin for WMMA; scalar FMA.
__global__ void fc_logits(const float* __restrict__ X,
                          const float* __restrict__ W,
                          const float* __restrict__ b,
                          float* __restrict__ out)
{
    const int idx = blockIdx.x * blockDim.x + threadIdx.x;
    if (idx >= 256 * 5) return;
    const int r = idx / 5, c = idx % 5;
    float s = b[c];
    const float* xr = X + (size_t)r * 256;
    #pragma unroll 8
    for (int k = 0; k < 256; ++k)
        s = fmaf(xr[k], W[k * 5 + c], s);
    out[idx] = s;
}

extern "C" void kernel_launch(void* const* d_in, const int* in_sizes, int n_in,
                              void* d_out, int out_size, void* d_ws, size_t ws_size,
                              hipStream_t stream) {
    (void)in_sizes; (void)n_in; (void)out_size; (void)ws_size;

    // setup_inputs() dict order:
    // 0:x 1:biobert_cls  2..9:g1*  10..17:g2*  18..25:g3*
    // 26:ca_wq 27:ca_bq 28:ca_wk 29:ca_bk 30:ca_wv 31:ca_bv 32:ca_wo 33:ca_bo
    // 34:ca_ln_g 35:ca_ln_b 36:ffn_w1 37:ffn_b1 38:ffn_w2 39:ffn_b2
    // 40:fc_w 41:fc_b 42:fn_g 43:fn_b 44:edge_src 45:edge_dst 46:batch
    const float* biobert = (const float*)d_in[1];
    const float* ca_wv   = (const float*)d_in[30];
    const float* ca_bv   = (const float*)d_in[31];
    const float* ca_wo   = (const float*)d_in[32];
    const float* ca_bo   = (const float*)d_in[33];
    const float* ca_ln_g = (const float*)d_in[34];
    const float* ca_ln_b = (const float*)d_in[35];
    const float* ffn_w1  = (const float*)d_in[36];
    const float* ffn_b1  = (const float*)d_in[37];
    const float* ffn_w2  = (const float*)d_in[38];
    const float* ffn_b2  = (const float*)d_in[39];
    const float* fc_w    = (const float*)d_in[40];
    const float* fc_b    = (const float*)d_in[41];
    const float* fn_g    = (const float*)d_in[42];
    const float* fn_b    = (const float*)d_in[43];

    float* ws  = (float*)d_ws;           // 6 x 256*256 f32 = 1.5 MB scratch
    float* V   = ws + 0 * 65536;
    float* P   = ws + 1 * 65536;
    float* CTX = ws + 2 * 65536;
    float* T   = ws + 3 * 65536;
    float* F   = ws + 4 * 65536;
    float* O   = ws + 5 * 65536;

    const dim3 wblk(32);                 // one wave per block; EXEC all-ones for WMMA
    const int tiles256 = (256 / 16) * (256 / 16);   // 256 tiles

    // ctx == V because softmax over a singleton key axis is exactly 1.0.
    wmma_gemm_f16<<<tiles256, wblk, 0, stream>>>(biobert, ca_wv, ca_bv, V, 256, 256, 768, 0);
    wmma_gemm_f16<<<tiles256, wblk, 0, stream>>>(V, ca_wo, ca_bo, P, 256, 256, 256, 0);
    layernorm256 <<<256, wblk, 0, stream>>>(P, nullptr, ca_ln_g, ca_ln_b, CTX);
    wmma_gemm_f16<<<tiles256, wblk, 0, stream>>>(CTX, ffn_w1, ffn_b1, T, 256, 256, 256, 1);
    wmma_gemm_f16<<<tiles256, wblk, 0, stream>>>(T, ffn_w2, ffn_b2, F, 256, 256, 256, 0);
    layernorm256 <<<256, wblk, 0, stream>>>(F, CTX, fn_g, fn_b, O);
    fc_logits    <<<(256 * 5 + 255) / 256, dim3(256), 0, stream>>>(O, fc_w, fc_b, (float*)d_out);
}
